// PrefixMultiHeadAttention_20289425506487
// MI455X (gfx1250) — matrix operand order repaired and checked by reading, not compiled, not run on hardware
//
#include <hip/hip_runtime.h>
#include <hip/hip_bf16.h>

typedef __attribute__((ext_vector_type(16))) __bf16 v16bf;
typedef __attribute__((ext_vector_type(8)))  __bf16 v8bf;
typedef __attribute__((ext_vector_type(8)))  float  v8f;

constexpr int Bc = 4, Sc = 2048, Dc = 768, Hc = 12, Pc = 64;
constexpr int LTc = Pc + Sc;   // 2112 total key positions
constexpr int Kc = Dc;         // GEMM reduction dim

__device__ __forceinline__ v16bf pack16(v8bf lo, v8bf hi) {
  v16bf r;
#pragma unroll
  for (int i = 0; i < 8; ++i) { r[i] = lo[i]; r[i + 8] = hi[i]; }
  return r;
}

__device__ __forceinline__ v8f wmma_bf16(v16bf a, v16bf b, v8f c) {
  return __builtin_amdgcn_wmma_f32_16x16x32_bf16(false, a, false, b, (short)0, c,
                                                 false, false);
}

// CDNA5 async global->LDS copy (ASYNCcnt-tracked, no VGPR round trip).
// Generic pointers to LDS carry the LDS offset in their low 32 bits.
__device__ __forceinline__ void async_load_b128_to_lds(void* ldsp, const void* gptr) {
  unsigned lds = (unsigned)(unsigned long long)ldsp;
  asm volatile("global_load_async_to_lds_b128 %0, %1, off"
               :
               : "v"(lds), "v"((unsigned long long)gptr)
               : "memory");
}
__device__ __forceinline__ void wait_asynccnt0() {
  asm volatile("s_wait_asynccnt 0x0" ::: "memory");
}

// ---------------------------------------------------------------- casts
__global__ void cast_bf16_kernel(const float* __restrict__ src,
                                 __bf16* __restrict__ dst, size_t n) {
  size_t i = (size_t)blockIdx.x * blockDim.x + threadIdx.x;
  size_t stride = (size_t)gridDim.x * blockDim.x;
  for (; i < n; i += stride) dst[i] = (__bf16)src[i];
}

// copy fp32 prefixes into bf16 KV caches at rows [0, P)
__global__ void prefix_kernel(const float* __restrict__ Pk,
                              const float* __restrict__ Pv,
                              __bf16* __restrict__ Kb, __bf16* __restrict__ Vb) {
  const size_t n = (size_t)Bc * Pc * Dc;
  size_t i = (size_t)blockIdx.x * blockDim.x + threadIdx.x;
  size_t stride = (size_t)gridDim.x * blockDim.x;
  for (; i < n; i += stride) {
    size_t b = i / ((size_t)Pc * Dc);
    size_t rem = i - b * ((size_t)Pc * Dc);
    size_t o = b * (size_t)LTc * Dc + rem;
    Kb[o] = (__bf16)Pk[i];
    Vb[o] = (__bf16)Pv[i];
  }
}

// ------------------------------------------------- NT GEMM: out = A @ W^T
// A: [M, K] bf16 row-major (M = B*S), W: [N, K] bf16 row-major (N = K = 768).
// Register-blocked: each wave computes a 64x32 output block (4x2 WMMA tiles),
// reusing each fragment across the other dimension (768 B loaded per WMMA).
// Output row remap: orow = (m / S) * rows_out + row_off + (m % S)
// F32OUT selects the store flavor at compile time (no per-store branching).
template <bool F32OUT>
__global__ __launch_bounds__(256) void gemm_nt_bf16_kernel(
    const __bf16* __restrict__ A, const __bf16* __restrict__ W,
    __bf16* __restrict__ outb, float* __restrict__ outf,
    int rows_out, int row_off) {
  const int wave = threadIdx.x >> 5;
  const int lane = threadIdx.x & 31;
  const int lh = lane >> 4;      // half-wave select
  const int lr = lane & 15;
  const int m0 = (blockIdx.y * 8 + wave) * 64;
  const int n0 = blockIdx.x * 32;

  v8f acc[4][2];
#pragma unroll
  for (int i = 0; i < 4; ++i)
#pragma unroll
    for (int j = 0; j < 2; ++j)
      acc[i][j] = v8f{0.f, 0.f, 0.f, 0.f, 0.f, 0.f, 0.f, 0.f};

  const __bf16* arow[4];
  const __bf16* wrow[2];
#pragma unroll
  for (int i = 0; i < 4; ++i) arow[i] = A + (size_t)(m0 + 16 * i + lr) * Kc;
#pragma unroll
  for (int j = 0; j < 2; ++j) wrow[j] = W + (size_t)(n0 + 16 * j + lr) * Kc;

  for (int k0 = 0; k0 < Kc; k0 += 32) {
    v16bf af[4], bfr[2];
#pragma unroll
    for (int i = 0; i < 4; ++i) {
      // A-layout (interleaved): lane<16 K {0..7,16..23}, lane>=16 {8..15,24..31}
      v8bf lo = *(const v8bf*)(arow[i] + k0 + lh * 8);
      v8bf hi = *(const v8bf*)(arow[i] + k0 + lh * 8 + 16);
      af[i] = pack16(lo, hi);
    }
#pragma unroll
    for (int j = 0; j < 2; ++j) {
      // B-layout (interleaved): lane<16 K {0..7,16..23}, lane>=16 {8..15,24..31}
      v8bf lo = *(const v8bf*)(wrow[j] + k0 + lh * 8);
      v8bf hi = *(const v8bf*)(wrow[j] + k0 + lh * 8 + 16);
      bfr[j] = pack16(lo, hi);
    }
#pragma unroll
    for (int i = 0; i < 4; ++i)
#pragma unroll
      for (int j = 0; j < 2; ++j)
        acc[i][j] = wmma_bf16(af[i], bfr[j], acc[i][j]);
  }

#pragma unroll
  for (int i = 0; i < 4; ++i)
#pragma unroll
    for (int j = 0; j < 2; ++j)
#pragma unroll
      for (int r = 0; r < 8; ++r) {
        int m = m0 + 16 * i + lh * 8 + r;            // C row = M
        int orow = (m >> 11) * rows_out + row_off + (m & (Sc - 1));
        size_t idx = (size_t)orow * Dc + n0 + 16 * j + lr;  // C col = N
        if constexpr (F32OUT) outf[idx] = acc[i][j][r];
        else                  outb[idx] = (__bf16)acc[i][j][r];
      }
}

// ------------------------------------------------- fused attention
// Grid: (S/128, B*H). Block: 256 thr = 8 waves; wave w owns q-tile q0 = x*128+16w.
// Two-pass softmax: pass 0 -> per-query running (max, sum); pass 1 -> normalized
// probs written to d_out and fed straight into the P*V WMMAs.
__global__ __launch_bounds__(256) void attn_kernel(
    const __bf16* __restrict__ Qb, const __bf16* __restrict__ Kb,
    const __bf16* __restrict__ Vb, const int* __restrict__ mask,
    float* __restrict__ attnw, __bf16* __restrict__ Ob) {
  __shared__ __bf16 ldsK[32][72];    // 32 keys x 64 head-dims (+pad)
  __shared__ __bf16 ldsVT[64][40];   // V transposed: [d][kpos] (+pad, 16B aligned rows)

  const int b = blockIdx.y / Hc;
  const int h = blockIdx.y % Hc;
  const int wave = threadIdx.x >> 5;
  const int lane = threadIdx.x & 31;
  const int lh = lane >> 4, lr = lane & 15;
  const int q0 = blockIdx.x * 128 + wave * 16;

  // Q as B-operand fragments (head-dim 0-31 and 32-63); lane = query column.
  const __bf16* qrow = Qb + (size_t)(b * Sc + q0 + lr) * Dc + h * 64;
  v16bf bq[2];
#pragma unroll
  for (int f = 0; f < 2; ++f) {
    v8bf lo = *(const v8bf*)(qrow + f * 32 + lh * 8);
    v8bf hi = *(const v8bf*)(qrow + f * 32 + lh * 8 + 16);
    bq[f] = pack16(lo, hi);
  }

  float mi = -3.0e38f, li = 0.f, inv_li = 0.f;
  v8f acc[4];
#pragma unroll
  for (int t = 0; t < 4; ++t) acc[t] = v8f{0.f, 0.f, 0.f, 0.f, 0.f, 0.f, 0.f, 0.f};

  const int ct = threadIdx.x >> 3;        // cooperative stage: row 0..31
  const int cc = (threadIdx.x & 7) * 8;   // col chunk 0..56

  for (int pass = 0; pass < 2; ++pass) {
    if (pass == 1) inv_li = 1.f / li;
    for (int kp0 = 0; kp0 < LTc; kp0 += 32) {
      __syncthreads();
      {  // stage 32x64 K tile via CDNA5 async global->LDS copy
        async_load_b128_to_lds(
            &ldsK[ct][cc],
            Kb + ((size_t)b * LTc + kp0 + ct) * Dc + h * 64 + cc);
        if (pass == 1) {  // V needs a transpose -> explicit staging
          v8bf vv =
              *(const v8bf*)(Vb + ((size_t)b * LTc + kp0 + ct) * Dc + h * 64 + cc);
#pragma unroll
          for (int j = 0; j < 8; ++j) ldsVT[cc + j][ct] = vv[j];
        }
        wait_asynccnt0();
      }
      __syncthreads();

      // scores^T: C rows = key pos (VGPR), cols = query (lane)
      float sv[16];
#pragma unroll
      for (int t = 0; t < 2; ++t) {
        v8f c = {0.f, 0.f, 0.f, 0.f, 0.f, 0.f, 0.f, 0.f};
#pragma unroll
        for (int f = 0; f < 2; ++f) {
          v8bf lo = *(const v8bf*)&ldsK[t * 16 + lr][f * 32 + lh * 8];
          v8bf hi = *(const v8bf*)&ldsK[t * 16 + lr][f * 32 + lh * 8 + 16];
          c = wmma_bf16(pack16(lo, hi), bq[f], c);
        }
#pragma unroll
        for (int r = 0; r < 8; ++r) {
          int kp = kp0 + t * 16 + lh * 8 + r;
          float madd = 0.f;
          if (kp >= Pc) madd = (mask[b * Sc + kp - Pc] == 0) ? -3.0e38f : 0.f;
          sv[t * 8 + r] = c[r] * 0.125f + madd;  // 1/sqrt(64)
        }
      }

      if (pass == 0) {
        float mloc = -3.0e38f;
#pragma unroll
        for (int i = 0; i < 16; ++i) mloc = fmaxf(mloc, sv[i]);
        mloc = fmaxf(mloc, __shfl_xor(mloc, 16, 32));  // merge lane pair (same q)
        float mnew = fmaxf(mi, mloc);
        float lst = 0.f;
#pragma unroll
        for (int i = 0; i < 16; ++i) lst += __expf(sv[i] - mnew);
        lst += __shfl_xor(lst, 16, 32);
        li = li * __expf(mi - mnew) + lst;
        mi = mnew;
      } else {
        float p[16];
#pragma unroll
        for (int i = 0; i < 16; ++i) p[i] = __expf(sv[i] - mi) * inv_li;

        // write normalized attention weights (8 contiguous kp per lane per tile)
        float* ap = attnw + ((size_t)(b * Hc + h) * Sc + q0 + lr) * LTc + kp0;
#pragma unroll
        for (int t = 0; t < 2; ++t) {
          float4* dst = (float4*)(ap + t * 16 + lh * 8);
          dst[0] = make_float4(p[t * 8 + 0], p[t * 8 + 1], p[t * 8 + 2], p[t * 8 + 3]);
          dst[1] = make_float4(p[t * 8 + 4], p[t * 8 + 5], p[t * 8 + 6], p[t * 8 + 7]);
        }

        // C-layout prob pair is exactly the 16-bit A-matrix interleaved layout.
        v16bf pf;
#pragma unroll
        for (int i = 0; i < 16; ++i) pf[i] = (__bf16)p[i];

        // O(16q x 64d) += P(16x32) * V(32x16 per d-tile); B = V from ldsVT rows.
#pragma unroll
        for (int dt = 0; dt < 4; ++dt) {
          v8bf lo = *(const v8bf*)&ldsVT[dt * 16 + lr][lh * 8];
          v8bf hi = *(const v8bf*)&ldsVT[dt * 16 + lr][lh * 8 + 16];
          acc[dt] = wmma_bf16(pf, pack16(lo, hi), acc[dt]);
        }
      }
    }
  }

  // acc: rows q = lh*8 + r, cols d = dt*16 + lr (already normalized)
#pragma unroll
  for (int dt = 0; dt < 4; ++dt)
#pragma unroll
    for (int r = 0; r < 8; ++r) {
      int q = q0 + lh * 8 + r;
      Ob[(size_t)(b * Sc + q) * Dc + h * 64 + dt * 16 + lr] = (__bf16)acc[dt][r];
    }
}

// ---------------------------------------------------------------- launcher
extern "C" void kernel_launch(void* const* d_in, const int* in_sizes, int n_in,
                              void* d_out, int out_size, void* d_ws, size_t ws_size,
                              hipStream_t stream) {
  const float* X  = (const float*)d_in[0];
  const float* Pk = (const float*)d_in[1];
  const float* Pv = (const float*)d_in[2];
  const int*  msk = (const int*)d_in[3];
  const float* Wq = (const float*)d_in[4];
  const float* Wk = (const float*)d_in[5];
  const float* Wv = (const float*)d_in[6];
  const float* Wo = (const float*)d_in[7];

  float* out_main = (float*)d_out;                       // [B,S,D] fp32
  float* attnw = out_main + (size_t)Bc * Sc * Dc;        // [B,H,S,P+S] fp32

  char* w = (char*)d_ws;
  auto alloc = [&](size_t bytes) -> char* {
    char* p = w;
    w += (bytes + 255) & ~(size_t)255;
    return p;
  };
  const size_t BSD = (size_t)Bc * Sc * Dc;
  const size_t DD  = (size_t)Dc * Dc;
  const size_t KVN = (size_t)Bc * LTc * Dc;

  __bf16* Xb  = (__bf16*)alloc(BSD * 2);
  __bf16* Wqb = (__bf16*)alloc(DD * 2);
  __bf16* Wkb = (__bf16*)alloc(DD * 2);
  __bf16* Wvb = (__bf16*)alloc(DD * 2);
  __bf16* Wob = (__bf16*)alloc(DD * 2);
  __bf16* Qb  = (__bf16*)alloc(BSD * 2);
  __bf16* Kb  = (__bf16*)alloc(KVN * 2);
  __bf16* Vb  = (__bf16*)alloc(KVN * 2);
  __bf16* Ob  = (__bf16*)alloc(BSD * 2);

  cast_bf16_kernel<<<2048, 256, 0, stream>>>(X, Xb, BSD);
  cast_bf16_kernel<<<1024, 256, 0, stream>>>(Wq, Wqb, DD);
  cast_bf16_kernel<<<1024, 256, 0, stream>>>(Wk, Wkb, DD);
  cast_bf16_kernel<<<1024, 256, 0, stream>>>(Wv, Wvb, DD);
  cast_bf16_kernel<<<1024, 256, 0, stream>>>(Wo, Wob, DD);
  prefix_kernel<<<768, 256, 0, stream>>>(Pk, Pv, Kb, Vb);

  dim3 gg(Dc / 32, (Bc * Sc) / 512);  // (24, 16): 64x32 per wave, 8 waves/block
  gemm_nt_bf16_kernel<false><<<gg, 256, 0, stream>>>(Xb, Wqb, Qb, nullptr, Sc, 0);
  gemm_nt_bf16_kernel<false><<<gg, 256, 0, stream>>>(Xb, Wkb, Kb, nullptr, LTc, Pc);
  gemm_nt_bf16_kernel<false><<<gg, 256, 0, stream>>>(Xb, Wvb, Vb, nullptr, LTc, Pc);

  attn_kernel<<<dim3(Sc / 128, Bc * Hc), 256, 0, stream>>>(Qb, Kb, Vb, msk,
                                                           attnw, Ob);

  gemm_nt_bf16_kernel<true><<<gg, 256, 0, stream>>>(Ob, Wob, nullptr, out_main, Sc, 0);
}
